// HGTConvLayer_68247030333846
// MI455X (gfx1250) — matrix operand order repaired and checked by reading, not compiled, not run on hardware
//
#include <hip/hip_runtime.h>
#include <stdint.h>

#define NUN 100000
#define NCN 50000
#define NE  200000
#define HIDN 256
#define NHH 8
#define SCALE_QK 0.17677669529663687f   // 32^-0.5
#define TPB_TILES 16                     // 16-row tiles per block (256 rows)
#define LDS_AROW 33                      // v8bf units per LDS A row (pad: bank-conflict-free)

typedef __bf16 bf16_t;
typedef __attribute__((ext_vector_type(16))) __bf16 v16bf;
typedef __attribute__((ext_vector_type(8)))  __bf16 v8bf;
typedef __attribute__((ext_vector_type(8)))  float  v8f;

// ---------- helpers ----------
__device__ __forceinline__ bf16_t f2bf(float f) {
  union { float f; uint32_t u; } in; in.f = f;
  uint32_t r = (in.u + 0x7FFFu + ((in.u >> 16) & 1u)) >> 16;   // RNE
  union { uint16_t u; bf16_t b; } out; out.u = (uint16_t)r;
  return out.b;
}
__device__ __forceinline__ uint32_t enc_f(float f) {
  uint32_t u = __float_as_uint(f);
  return (u & 0x80000000u) ? ~u : (u | 0x80000000u);
}
__device__ __forceinline__ float dec_f(uint32_t k) {
  uint32_t u = (k & 0x80000000u) ? (k & 0x7FFFFFFFu) : ~k;
  return __uint_as_float(u);
}

// ---------- persistent-B / LDS-staged-A GEMM building blocks ----------
// B fragments (16x16 N-strip, all K=256) held in registers: 8 x v16bf = 64 VGPRs.
// B[k][n] = W[n][k] (torch Linear: C = A @ W^T), W row-major bf16.
__device__ __forceinline__ void load_bfrag(const bf16_t* __restrict__ W, int n0, v16bf* bfr) {
  const int lane = threadIdx.x & 31;
  const int n = lane & 15, hi = lane >> 4;
  const bf16_t* brow = W + (size_t)(n0 + n) * HIDN + hi * 16;
#pragma unroll
  for (int ks = 0; ks < 8; ++ks) bfr[ks] = *(const v16bf*)(brow + ks * 32);
}

// stage one 16x256 bf16 A tile into LDS: 512 threads, one 16B chunk each (wave w -> row w)
__device__ __forceinline__ void stage_a(const bf16_t* __restrict__ A, int row0, v8bf* dst) {
  const int t = threadIdx.x;
  const int row = t >> 5, seg = t & 31;
  dst[row * LDS_AROW + seg] = *(const v8bf*)(A + (size_t)(row0 + row) * HIDN + seg * 8);
}

// one 16x16 WMMA tile from LDS A + register B
__device__ __forceinline__ void compute_tile(const v8bf* __restrict__ lds, const v16bf* bfr, v8f& c) {
  const int lane = threadIdx.x & 31;
  const int m = lane & 15, hi = lane >> 4;
#pragma unroll
  for (int i = 0; i < 8; ++i) c[i] = 0.f;
#pragma unroll
  for (int ks = 0; ks < 8; ++ks) {
    union { v16bf v; v8bf h[2]; } a;
    a.h[0] = lds[m * LDS_AROW + ks * 4 + 2 * hi];       // K = ks*32 + hi*16 + 0..7
    a.h[1] = lds[m * LDS_AROW + ks * 4 + 2 * hi + 1];   // K = ks*32 + hi*16 + 8..15
    c = __builtin_amdgcn_wmma_f32_16x16x32_bf16(false, a.v, false, bfr[ks], (short)0, c, false, false);
  }
}

// ---------- GEMM kernels (512 threads = 16 waves; wave w owns cols [16w,16w+16)) ----------
// C_bf16[N,256] = A_bf16 @ W^T   (node Q/K/V projections)
__global__ __launch_bounds__(512) void proj_gemm(const bf16_t* __restrict__ A,
                                                 const bf16_t* __restrict__ W,
                                                 bf16_t* __restrict__ C, int ntiles) {
  __shared__ v8bf atile[2][16 * LDS_AROW];
  const int n0 = (threadIdx.x >> 5) * 16;
  v16bf bfr[8];
  load_bfrag(W, n0, bfr);
  const int t0 = blockIdx.x * TPB_TILES;
  const int tend = (t0 + TPB_TILES < ntiles) ? t0 + TPB_TILES : ntiles;
  stage_a(A, t0 * 16, atile[0]);
  __syncthreads();
  int buf = 0;
  for (int ti = t0; ti < tend; ++ti) {
    if (ti + 1 < tend) stage_a(A, (ti + 1) * 16, atile[buf ^ 1]);
    v8f c;
    compute_tile(atile[buf], bfr, c);
    const int lane = threadIdx.x & 31;
    const int n = lane & 15, mb = (lane >> 4) * 8;
#pragma unroll
    for (int r = 0; r < 8; ++r)
      C[(size_t)(ti * 16 + mb + r) * HIDN + n0 + n] = f2bf(c[r]);
    __syncthreads();
    buf ^= 1;
  }
}

// comb (=|+=) scale * ( aggWV_bf16 @ mw^T + deg[row]*mb[col] )
__global__ __launch_bounds__(512) void msg_gemm(const bf16_t* __restrict__ AGGB,
                                                const bf16_t* __restrict__ MW,
                                                const float* __restrict__ mbias,
                                                const float* __restrict__ deg,
                                                const float* __restrict__ imp1,
                                                const float* __restrict__ imp2,
                                                const float* __restrict__ imp3,
                                                int sel, int beta,
                                                float* __restrict__ comb, int ntiles) {
  __shared__ v8bf atile[2][16 * LDS_AROW];
  const int n0 = (threadIdx.x >> 5) * 16;
  v16bf bfr[8];
  load_bfrag(MW, n0, bfr);
  float scale;
  {
    float r1 = 1.f / (1.f + __expf(-imp1[0]));
    float r2 = 1.f / (1.f + __expf(-imp2[0]));
    float r3 = 1.f / (1.f + __expf(-imp3[0]));
    if (sel == 0) scale = r1;                        // softmax over single rel == 1
    else {
      float e2 = __expf(r2), e3 = __expf(r3), d = e2 + e3;
      scale = (sel == 1) ? r2 * (e2 / d) : r3 * (e3 / d);
    }
  }
  const int lane = threadIdx.x & 31;
  const int n = lane & 15, mb = (lane >> 4) * 8;
  const float bc = mbias[n0 + n];
  const int t0 = blockIdx.x * TPB_TILES;
  const int tend = (t0 + TPB_TILES < ntiles) ? t0 + TPB_TILES : ntiles;
  stage_a(AGGB, t0 * 16, atile[0]);
  __syncthreads();
  int buf = 0;
  for (int ti = t0; ti < tend; ++ti) {
    if (ti + 1 < tend) stage_a(AGGB, (ti + 1) * 16, atile[buf ^ 1]);
    v8f c;
    compute_tile(atile[buf], bfr, c);
#pragma unroll
    for (int r = 0; r < 8; ++r) {
      int row = ti * 16 + mb + r;
      float v = (c[r] + deg[row] * bc) * scale;
      size_t idx = (size_t)row * HIDN + n0 + n;
      if (beta) comb[idx] += v; else comb[idx] = v;
    }
    __syncthreads();
    buf ^= 1;
  }
}

// out = LayerNorm( H + comb_bf16 @ pw^T + pb ) * g + b   (fused GEMM+residual+LN)
__global__ __launch_bounds__(512) void out_gemm_ln(const bf16_t* __restrict__ COMBB,
                                                   const bf16_t* __restrict__ PW,
                                                   const float* __restrict__ pb,
                                                   const float* __restrict__ H,
                                                   const float* __restrict__ g,
                                                   const float* __restrict__ bia,
                                                   float* __restrict__ out, int ntiles) {
  __shared__ v8bf atile[2][16 * LDS_AROW];
  __shared__ float ot[16 * 264];                    // padded row stride
  const int wave = threadIdx.x >> 5;
  const int n0 = wave * 16;
  v16bf bfr[8];
  load_bfrag(PW, n0, bfr);
  const int lane = threadIdx.x & 31;
  const int n = lane & 15, mb = (lane >> 4) * 8;
  const int t0 = blockIdx.x * TPB_TILES;
  const int tend = (t0 + TPB_TILES < ntiles) ? t0 + TPB_TILES : ntiles;
  stage_a(COMBB, t0 * 16, atile[0]);
  __syncthreads();
  int buf = 0;
  for (int ti = t0; ti < tend; ++ti) {
    if (ti + 1 < tend) stage_a(COMBB, (ti + 1) * 16, atile[buf ^ 1]);
    v8f c;
    compute_tile(atile[buf], bfr, c);
#pragma unroll
    for (int r = 0; r < 8; ++r) ot[(mb + r) * 264 + n0 + n] = c[r];
    __syncthreads();
    // LayerNorm: wave w handles row w; 32 lanes x 8 cols
    const int grow = ti * 16 + wave;
    float vals[8], s = 0.f, s2 = 0.f;
#pragma unroll
    for (int j = 0; j < 8; ++j) {
      int cix = lane * 8 + j;
      float v = ot[wave * 264 + cix] + H[(size_t)grow * HIDN + cix] + pb[cix];
      vals[j] = v; s += v; s2 += v * v;
    }
#pragma unroll
    for (int sft = 1; sft < 32; sft <<= 1) {
      s  += __shfl_xor(s,  sft, 32);
      s2 += __shfl_xor(s2, sft, 32);
    }
    float mean = s * (1.f / HIDN);
    float var  = s2 * (1.f / HIDN) - mean * mean;
    float rstd = rsqrtf(var + 1e-5f);
#pragma unroll
    for (int j = 0; j < 8; ++j) {
      int cix = lane * 8 + j;
      out[(size_t)grow * HIDN + cix] = (vals[j] - mean) * rstd * g[cix] + bia[cix];
    }
    __syncthreads();
    buf ^= 1;
  }
}

// ---------- edge-phase kernels (memory-bound; Q/K/V in bf16) ----------
__global__ __launch_bounds__(256) void edge_scores(const bf16_t* __restrict__ q,
                                                   const bf16_t* __restrict__ k,
                                                   const int* __restrict__ src,
                                                   const int* __restrict__ dst,
                                                   const float* __restrict__ ew,
                                                   float* __restrict__ sc,
                                                   uint32_t* __restrict__ smax, int ne) {
  int e = blockIdx.x * 8 + (threadIdx.x >> 5);
  if (e >= ne) return;
  const int lane = threadIdx.x & 31;
  const int h = lane >> 2, sub = lane & 3;
  const v8bf qv = *(const v8bf*)(q + (size_t)dst[e] * HIDN + h * 32 + sub * 8);
  const v8bf kv = *(const v8bf*)(k + (size_t)src[e] * HIDN + h * 32 + sub * 8);
  float p = 0.f;
#pragma unroll
  for (int j = 0; j < 8; ++j) p += (float)qv[j] * (float)kv[j];
  p += __shfl_xor(p, 1, 32);
  p += __shfl_xor(p, 2, 32);
  if (sub == 0) {
    float s = p * SCALE_QK * ew[e];
    sc[(size_t)e * NHH + h] = s;
    atomicMax(&smax[(size_t)dst[e] * NHH + h], enc_f(s));
  }
}

__global__ __launch_bounds__(256) void edge_exp(const int* __restrict__ dst,
                                                float* __restrict__ sc_ex,
                                                const uint32_t* __restrict__ smax,
                                                float* __restrict__ ssum,
                                                float* __restrict__ deg, int ne) {
  long long i = (long long)blockIdx.x * blockDim.x + threadIdx.x;
  if (i >= (long long)ne * NHH) return;
  int e = (int)(i >> 3), h = (int)(i & 7);
  int d = dst[e];
  float m = dec_f(smax[(size_t)d * NHH + h]);
  float ex = __expf(sc_ex[i] - m);
  sc_ex[i] = ex;
  atomicAdd(&ssum[(size_t)d * NHH + h], ex);
  if (h == 0) atomicAdd(&deg[d], 1.0f);
}

__global__ __launch_bounds__(256) void edge_scatter(const bf16_t* __restrict__ v,
                                                    const int* __restrict__ src,
                                                    const int* __restrict__ dst,
                                                    const float* __restrict__ ex,
                                                    const float* __restrict__ ssum,
                                                    float* __restrict__ agg) {
  const int e = blockIdx.x;
  const int c = threadIdx.x;
  const int h = c >> 5;
  const int d = dst[e];
  float p = ex[(size_t)e * NHH + h] / (ssum[(size_t)d * NHH + h] + 1e-10f);
  atomicAdd(&agg[(size_t)d * HIDN + c], (float)v[(size_t)src[e] * HIDN + c] * p);
}

// ---------- utility kernels ----------
__global__ void fill_u32(uint32_t* p, uint32_t v, long long n) {
  long long i = (long long)blockIdx.x * blockDim.x + threadIdx.x;
  if (i < n) p[i] = v;
}
__global__ void conv_bf(const float* __restrict__ s, bf16_t* __restrict__ d, long long n) {
  long long i = (long long)blockIdx.x * blockDim.x + threadIdx.x;
  if (i < n) d[i] = f2bf(s[i]);
}

// ---------- host launch ----------
extern "C" void kernel_launch(void* const* d_in, const int* in_sizes, int n_in,
                              void* d_out, int out_size, void* d_ws, size_t ws_size,
                              hipStream_t stream) {
  (void)in_sizes; (void)n_in; (void)out_size; (void)ws_size;
  const float* h_user = (const float*)d_in[0];
  const float* h_comp = (const float*)d_in[1];
  const int*   srcI[3] = {(const int*)d_in[2], (const int*)d_in[5], (const int*)d_in[8]};
  const int*   dstI[3] = {(const int*)d_in[3], (const int*)d_in[6], (const int*)d_in[9]};
  const float* ewP[3]  = {(const float*)d_in[4], (const float*)d_in[7], (const float*)d_in[10]};
  const float* mbP[3]  = {(const float*)d_in[15], (const float*)d_in[21], (const float*)d_in[27]};
  const float* imp1 = (const float*)d_in[16];
  const float* imp2 = (const float*)d_in[22];
  const float* imp3 = (const float*)d_in[28];

  const size_t MATE   = (size_t)HIDN * HIDN;             // 65536 elems
  const size_t NODE_F = (size_t)NUN * HIDN * 4;
  const size_t NODE_B = (size_t)NUN * HIDN * 2;
  char* ws = (char*)d_ws;
  bf16_t* WB = (bf16_t*)ws;                              // 14 bf16 weight matrices
  size_t off = (14 * MATE * 2 + 255) & ~(size_t)255;
  bf16_t*   HUB  = (bf16_t*)(ws + off);  off += NODE_B;
  bf16_t*   HCB  = (bf16_t*)(ws + off);  off += (size_t)NCN * HIDN * 2 + 256;
  bf16_t*   Qb   = (bf16_t*)(ws + off);  off += NODE_B;
  bf16_t*   Kbb  = (bf16_t*)(ws + off);  off += NODE_B;
  bf16_t*   Vbb  = (bf16_t*)(ws + off);  off += NODE_B;
  bf16_t*   AGGB = (bf16_t*)(ws + off);  off += NODE_B;  // bf16 scratch (agg/comb)
  float*    AGG  = (float*)(ws + off);   off += NODE_F;
  float*    EXb  = (float*)(ws + off);   off += ((size_t)NE * NHH * 4 + 255) & ~(size_t)255;
  uint32_t* SMAX = (uint32_t*)(ws + off); off += ((size_t)NUN * NHH * 4 + 255) & ~(size_t)255;
  float*    SSUM = (float*)(ws + off);   off += ((size_t)NUN * NHH * 4 + 255) & ~(size_t)255;
  float*    DEG  = (float*)(ws + off);   off += ((size_t)NUN * 4 + 255) & ~(size_t)255;
  float*    CU   = (float*)(ws + off);   off += NODE_F;
  float*    CC   = (float*)(ws + off);   off += (size_t)NCN * HIDN * 4;

  // bf16 weights: slots [4r+0..3]=qw,kw,vw,mw; 12=pw_user; 13=pw_comp
  const int widx[14] = {11,12,13,14, 17,18,19,20, 23,24,25,26, 29,33};
  for (int i = 0; i < 14; ++i)
    conv_bf<<<(int)(MATE / 256), 256, 0, stream>>>((const float*)d_in[widx[i]],
                                                   WB + (size_t)i * MATE, (long long)MATE);
  conv_bf<<<(int)((size_t)NUN * HIDN / 256), 256, 0, stream>>>(h_user, HUB, (long long)NUN * HIDN);
  conv_bf<<<(int)((size_t)NCN * HIDN / 256), 256, 0, stream>>>(h_comp, HCB, (long long)NCN * HIDN);

  const bf16_t* hdst[3] = {HCB, HUB, HUB};
  const bf16_t* hsrc[3] = {HUB, HCB, HUB};
  const int     ndst[3] = {NCN, NUN, NUN};
  const int     nsrc[3] = {NUN, NCN, NUN};
  float*        comb[3] = {CC, CU, CU};
  const int     betaA[3] = {0, 0, 1};
  auto gblk = [](int ntiles) { return (ntiles + TPB_TILES - 1) / TPB_TILES; };

  for (int r = 0; r < 3; ++r) {
    const bf16_t* qwB = WB + (size_t)(4 * r + 0) * MATE;
    const bf16_t* kwB = WB + (size_t)(4 * r + 1) * MATE;
    const bf16_t* vwB = WB + (size_t)(4 * r + 2) * MATE;
    const bf16_t* mwB = WB + (size_t)(4 * r + 3) * MATE;
    const int tdst = ndst[r] / 16, tsrc = nsrc[r] / 16;

    proj_gemm<<<gblk(tdst), 512, 0, stream>>>(hdst[r], qwB, Qb, tdst);
    proj_gemm<<<gblk(tsrc), 512, 0, stream>>>(hsrc[r], kwB, Kbb, tsrc);
    proj_gemm<<<gblk(tsrc), 512, 0, stream>>>(hsrc[r], vwB, Vbb, tsrc);

    long long nagg = (long long)ndst[r] * HIDN;
    long long nh8  = (long long)ndst[r] * NHH;
    fill_u32<<<(int)((nagg + 255) / 256), 256, 0, stream>>>((uint32_t*)AGG, 0u, nagg);
    fill_u32<<<(int)((nh8 + 255) / 256), 256, 0, stream>>>(SMAX, 0u, nh8);   // 0 < enc(-inf)
    fill_u32<<<(int)((nh8 + 255) / 256), 256, 0, stream>>>((uint32_t*)SSUM, 0u, nh8);
    fill_u32<<<(ndst[r] + 255) / 256, 256, 0, stream>>>((uint32_t*)DEG, 0u, ndst[r]);

    edge_scores<<<(NE + 7) / 8, 256, 0, stream>>>(Qb, Kbb, srcI[r], dstI[r], ewP[r], EXb, SMAX, NE);
    edge_exp<<<(int)(((long long)NE * NHH + 255) / 256), 256, 0, stream>>>(dstI[r], EXb, SMAX, SSUM, DEG, NE);
    edge_scatter<<<NE, 256, 0, stream>>>(Vbb, srcI[r], dstI[r], EXb, SSUM, AGG);

    conv_bf<<<(int)((nagg + 255) / 256), 256, 0, stream>>>(AGG, AGGB, nagg);
    msg_gemm<<<gblk(tdst), 512, 0, stream>>>(AGGB, mwB, mbP[r], DEG,
                                             imp1, imp2, imp3, r, betaA[r], comb[r], tdst);
  }

  conv_bf<<<(int)((size_t)NUN * HIDN / 256), 256, 0, stream>>>(CU, AGGB, (long long)NUN * HIDN);
  out_gemm_ln<<<gblk(NUN / 16), 512, 0, stream>>>(AGGB, WB + 12 * MATE, (const float*)d_in[30],
                                                  h_user, (const float*)d_in[31], (const float*)d_in[32],
                                                  (float*)d_out, NUN / 16);
  conv_bf<<<(int)((size_t)NCN * HIDN / 256), 256, 0, stream>>>(CC, AGGB, (long long)NCN * HIDN);
  out_gemm_ln<<<gblk(NCN / 16), 512, 0, stream>>>(AGGB, WB + 13 * MATE, (const float*)d_in[34],
                                                  h_comp, (const float*)d_in[35], (const float*)d_in[36],
                                                  (float*)d_out + (size_t)NUN * HIDN, NCN / 16);
}
